// SlowMAFTransform_85822036509175
// MI455X (gfx1250) — matrix operand order repaired and checked by reading, not compiled, not run on hardware
//
#include <hip/hip_runtime.h>

#define DIMX 64
#define KDIM 63
#define NHID 24
#define SLOPE 0.2f
#define NT 4                    // 16-row M tiles per wave

typedef __attribute__((ext_vector_type(16))) _Float16 v16h;
typedef __attribute__((ext_vector_type(8)))  float    v8f;

union Frag16 {
    v16h  v;
    uint4 u[2];
    _Float16 h[16];
};

__device__ __forceinline__ v8f wmma_f16(v16h a, v16h b, v8f c) {
    // D = A(16x32 f16) * B(32x16 f16) + C(16x16 f32)
    return __builtin_amdgcn_wmma_f32_16x16x32_f16(
        /*neg_a=*/false, a, /*neg_b=*/false, b,
        /*c_mod=*/(short)0, c, /*reuse_a=*/false, /*reuse_b=*/false);
}

// Hidden-channel storage permutation: channel n (0..31) lives at position
// p = 2*(n&15) | (n>>4); inverse chan(p) = ((p&1)<<4) | (p>>1).
// This makes each lane's two output channels (col, col+16) adjacent in LDS.
__device__ __forceinline__ int chan_of_pos(int p) { return ((p & 1) << 4) | (p >> 1); }

// A-operand (16x32, f16) from a row-major [16][32] f16 LDS tile (K = position index).
// lane<16:  row=lane,    K 0..7  in h[0..7], K 16..23 in h[8..15]
// lane>=16: row=lane-16, K 8..15 in h[0..7], K 24..31 in h[8..15]
__device__ __forceinline__ v16h load_a_lds(const _Float16* buf, int lane) {
    const int r = lane & 15;
    const int g = lane >> 4;
    Frag16 f;
    f.u[0] = *reinterpret_cast<const uint4*>(buf + r * 32 + g * 8);
    f.u[1] = *reinterpret_cast<const uint4*>(buf + r * 32 + 16 + g * 8);
    return f.v;
}

// B-operand (32x16, f16) from wT stored as [ncol][kpad] f16 (K contiguous per column).
// lane<16: col=lane, K 0..7,16..23 ; lane>=16: col=lane-16, K 8..15,24..31  (within chunk kc)
__device__ __forceinline__ v16h load_b_lds(const _Float16* wT, int kpad, int kc,
                                           int ntile, int lane) {
    const int c = lane & 15;
    const int g = lane >> 4;
    const _Float16* p = wT + (ntile * 16 + c) * kpad + kc * 32 + g * 8;
    Frag16 f;
    f.u[0] = *reinterpret_cast<const uint4*>(p);
    f.u[1] = *reinterpret_cast<const uint4*>(p + 16);
    return f.v;
}

// C layout: element (row = j + 8*(lane>=16), col = lane&15) in c[j].
// Bias + leaky relu, then store channels (col, col+16) as one packed f16 pair
// at positions (2*col, 2*col+1) -> a single ds_store_b32 per row.
__device__ __forceinline__ void act_store(_Float16* hb, v8f c0, v8f c1,
                                          const float* bias, int lane) {
    const int col   = lane & 15;
    const int rbase = (lane >> 4) * 8;
    const float bA = bias[col];
    const float bB = bias[16 + col];
#pragma unroll
    for (int j = 0; j < 8; ++j) {
        float v0 = c0[j] + bA;
        float v1 = c1[j] + bB;
        v0 = v0 > 0.f ? v0 : v0 * SLOPE;
        v1 = v1 > 0.f ? v1 : v1 * SLOPE;
        union { _Float16 h[2]; unsigned u; } pk;
        pk.h[0] = (_Float16)v0;
        pk.h[1] = (_Float16)v1;
        *reinterpret_cast<unsigned*>(hb + (rbase + j) * 32 + col * 2) = pk.u;
    }
}

__device__ __forceinline__ void cvt8(_Float16* dst, float4 a, float4 b) {
    dst[0] = (_Float16)a.x; dst[1] = (_Float16)a.y;
    dst[2] = (_Float16)a.z; dst[3] = (_Float16)a.w;
    dst[4] = (_Float16)b.x; dst[5] = (_Float16)b.y;
    dst[6] = (_Float16)b.z; dst[7] = (_Float16)b.w;
}

__global__ __launch_bounds__(256)
void maf_mlp_kernel(const float* __restrict__ x,
                    const float* __restrict__ W1, const float* __restrict__ b1,
                    const float* __restrict__ W2, const float* __restrict__ b2,
                    const float* __restrict__ W3, const float* __restrict__ b3,
                    const float* __restrict__ W4, const float* __restrict__ b4,
                    float* __restrict__ z, float* __restrict__ sws, int Btot) {
    __shared__ _Float16 w1T[32 * 64];        // [n][d] : layer-1 B, mask applied (natural K)
    __shared__ _Float16 w2T[32 * 32];        // [m][pos] : K-rows in permuted order
    __shared__ _Float16 w3T[32 * 32];        // [m][pos]
    __shared__ _Float16 w4T[16 * 32];        // [o][pos]
    __shared__ float bs1[32], bs2[32], bs3[32], bs4[2];
    __shared__ _Float16 hbuf[8][NT * 16 * 32];   // NT h-tiles per wave

    const int tid = threadIdx.x;
    const int k   = blockIdx.y;              // MLP index, serves output dim k+1

    // ---- stage weights (f32 -> f16), transpose, mask, permute K, zero-pad ----
    for (int i = tid; i < 32 * 64; i += 256) {
        const int n = i >> 6, d = i & 63;
        float v = (n < NHID && d <= k) ? W1[((size_t)k * NHID + n) * KDIM + d] : 0.f;
        w1T[i] = (_Float16)v;
    }
    for (int i = tid; i < 32 * 32; i += 256) {
        const int m = i >> 5, n = chan_of_pos(i & 31);
        const bool ok = (m < NHID) && (n < NHID);
        w2T[i] = (_Float16)(ok ? W2[((size_t)k * NHID + n) * NHID + m] : 0.f);
        w3T[i] = (_Float16)(ok ? W3[((size_t)k * NHID + n) * NHID + m] : 0.f);
    }
    for (int i = tid; i < 16 * 32; i += 256) {
        const int o = i >> 5, n = chan_of_pos(i & 31);
        float v = (o < 2 && n < NHID) ? W4[((size_t)k * NHID + n) * 2 + o] : 0.f;
        w4T[i] = (_Float16)v;
    }
    if (tid < 32) {
        bs1[tid] = (tid < NHID) ? b1[k * NHID + tid] : 0.f;
        bs2[tid] = (tid < NHID) ? b2[k * NHID + tid] : 0.f;
        bs3[tid] = (tid < NHID) ? b3[k * NHID + tid] : 0.f;
    }
    if (tid < 2) bs4[tid] = b4[k * 2 + tid];
    __syncthreads();

    const int wave = tid >> 5;
    const int lane = tid & 31;
    const int r0   = blockIdx.x * (128 * NT) + wave * (16 * NT);  // 16*NT rows/wave
    const int r    = lane & 15;
    const int g    = lane >> 4;
    _Float16* hb   = hbuf[wave];

    // ---- layer 1: B fragments held in registers, NT M-tiles swept ----
    {
        const v16h B00 = load_b_lds(w1T, 64, 0, 0, lane);
        const v16h B10 = load_b_lds(w1T, 64, 1, 0, lane);
        const v16h B01 = load_b_lds(w1T, 64, 0, 1, lane);
        const v16h B11 = load_b_lds(w1T, 64, 1, 1, lane);
#pragma unroll
        for (int t = 0; t < NT; ++t) {
            Frag16 alo, ahi;   // x K-chunks 0..31 / 32..63 in A layout
            const float4* xr4 =
                reinterpret_cast<const float4*>(x + (size_t)(r0 + t * 16 + r) * DIMX);
            cvt8(&alo.h[0], xr4[g * 2 + 0],      xr4[g * 2 + 1]);
            cvt8(&alo.h[8], xr4[4 + g * 2 + 0],  xr4[4 + g * 2 + 1]);
            cvt8(&ahi.h[0], xr4[8 + g * 2 + 0],  xr4[8 + g * 2 + 1]);
            cvt8(&ahi.h[8], xr4[12 + g * 2 + 0], xr4[12 + g * 2 + 1]);

            v8f c0 = {};
            v8f c1 = {};
            c0 = wmma_f16(alo.v, B00, c0);
            c0 = wmma_f16(ahi.v, B10, c0);
            c1 = wmma_f16(alo.v, B01, c1);
            c1 = wmma_f16(ahi.v, B11, c1);
            act_store(hb + t * 512, c0, c1, bs1, lane);
        }
    }
    // ---- layer 2 ----
    {
        const v16h B0 = load_b_lds(w2T, 32, 0, 0, lane);
        const v16h B1 = load_b_lds(w2T, 32, 0, 1, lane);
#pragma unroll
        for (int t = 0; t < NT; ++t) {
            v16h a = load_a_lds(hb + t * 512, lane);
            v8f d0 = {};
            v8f d1 = {};
            d0 = wmma_f16(a, B0, d0);
            d1 = wmma_f16(a, B1, d1);
            act_store(hb + t * 512, d0, d1, bs2, lane);
        }
    }
    // ---- layer 3 ----
    {
        const v16h B0 = load_b_lds(w3T, 32, 0, 0, lane);
        const v16h B1 = load_b_lds(w3T, 32, 0, 1, lane);
#pragma unroll
        for (int t = 0; t < NT; ++t) {
            v16h a = load_a_lds(hb + t * 512, lane);
            v8f d0 = {};
            v8f d1 = {};
            d0 = wmma_f16(a, B0, d0);
            d1 = wmma_f16(a, B1, d1);
            act_store(hb + t * 512, d0, d1, bs3, lane);
        }
    }
    // ---- layer 4: (s,t) land in C columns 0 and 1 ----
    {
        const v16h B0  = load_b_lds(w4T, 32, 0, 0, lane);
        const int col  = lane & 15;
        const float bv0 = bs4[0];
        const float bv1 = bs4[1];
#pragma unroll
        for (int t = 0; t < NT; ++t) {
            v16h a = load_a_lds(hb + t * 512, lane);
            v8f cst = {};
            cst = wmma_f16(a, B0, cst);
#pragma unroll
            for (int j = 0; j < 8; ++j) {
                float sj = cst[j] + bv0;              // meaningful on col==0 lanes
                float tj = cst[j] + bv1;              // meaningful on col==1 lanes
                float tn = __shfl_xor(tj, 1, 32);     // col0 lane grabs its t
                if (col == 0) {
                    const int row = r0 + t * 16 + (lane >> 4) * 8 + j;
                    const float s = sj;
                    const size_t zi = (size_t)row * DIMX + (k + 1);
                    z[zi] = x[zi] * expf(s) + tn;
                    sws[(size_t)k * Btot + row] = s;  // [k][b] layout
                }
            }
        }
    }
}

// dim-0 LeafParam + log-det reduction
__global__ __launch_bounds__(256)
void maf_finish_kernel(const float* __restrict__ x, const float* __restrict__ p0,
                       const float* __restrict__ sws, float* __restrict__ out, int Btot) {
    const int b = blockIdx.x * blockDim.x + threadIdx.x;
    if (b >= Btot) return;
    const float s0 = p0[0];
    const float t0 = p0[1];
    out[(size_t)b * DIMX] = x[(size_t)b * DIMX] * expf(s0) + t0;
    float sum = s0;
    for (int kk = 0; kk < KDIM; ++kk)
        sum += sws[(size_t)kk * Btot + b];            // coalesced across threads
    out[(size_t)Btot * DIMX + b] = -sum;
}

extern "C" void kernel_launch(void* const* d_in, const int* in_sizes, int n_in,
                              void* d_out, int out_size, void* d_ws, size_t ws_size,
                              hipStream_t stream) {
    (void)n_in; (void)out_size; (void)ws_size;
    const float* x  = (const float*)d_in[0];
    const float* p0 = (const float*)d_in[1];
    const float* W1 = (const float*)d_in[2];
    const float* b1 = (const float*)d_in[3];
    const float* W2 = (const float*)d_in[4];
    const float* b2 = (const float*)d_in[5];
    const float* W3 = (const float*)d_in[6];
    const float* b3 = (const float*)d_in[7];
    const float* W4 = (const float*)d_in[8];
    const float* b4 = (const float*)d_in[9];

    const int Btot = in_sizes[0] / DIMX;   // 32768
    float* z   = (float*)d_out;            // [B,64] then [B] of -log_det
    float* sws = (float*)d_ws;             // [KDIM][B] f32 scratch for s

    dim3 grid(Btot / (128 * NT), KDIM);    // 512 batch rows per block (8 waves x 64)
    maf_mlp_kernel<<<grid, 256, 0, stream>>>(x, W1, b1, W2, b2, W3, b3, W4, b4,
                                             z, sws, Btot);
    maf_finish_kernel<<<(Btot + 255) / 256, 256, 0, stream>>>(x, p0, sws, z, Btot);
}
